// SLSTM_30451318129076
// MI455X (gfx1250) — matrix operand order repaired and checked by reading, not compiled, not run on hardware
//
#include <hip/hip_runtime.h>
#include <hip/hip_bf16.h>

typedef __attribute__((ext_vector_type(16))) __bf16 v16bf;
typedef __attribute__((ext_vector_type(8)))  float  v8f;

#define DI __device__ __forceinline__

DI float sigf(float x) { return 1.f / (1.f + __expf(-x)); }

// ---- WMMA fragment loaders (wave32, 16x16x32 bf16) --------------------------
// A 16x32 bf16: lane m=lane&15 holds row m; lanes<16: K 0..7,16..23; lanes>=16: K 8..15,24..31
DI v16bf load_a(const __bf16* rowk, int lane) {
  union { v16bf v; unsigned int u[8]; } f;
  const int kh = (lane >> 4) * 8;
  #pragma unroll
  for (int j = 0; j < 8; ++j) {
    const int kb = ((j < 4) ? 0 : 16) + kh + (j & 3) * 2;
    f.u[j] = *(const unsigned int*)(rowk + kb);
  }
  return f.v;
}
// B prepacked: 32 lanes x 16 bf16 contiguous per (tile, ktile)
DI v16bf load_b(const __bf16* p) {
  union { v16bf v; uint4 q[2]; } f;
  const uint4* q = (const uint4*)p;
  f.q[0] = q[0]; f.q[1] = q[1];
  return f.v;
}

// ---- one-time weight packing -----------------------------------------------
// layout: [(k*16+nt)*KT + kt] * 512 + lane*16 + i ; value = W[k, nt*16+(lane&15), kt*32+(i>>3)*16+(lane>>4)*8+(i&7)]
__global__ void k_pack_wep(const float* __restrict__ W, __bf16* __restrict__ P) {
  int idx = blockIdx.x * 256 + threadIdx.x;
  if (idx >= 7 * 16 * 24 * 512) return;
  int i = idx & 15, lane = (idx >> 4) & 31;
  int t = idx >> 9;
  int kt = t % 24, tc = t / 24, nt = tc & 15, k = tc >> 4;
  int d = kt * 32 + (i >> 3) * 16 + (lane >> 4) * 8 + (i & 7);
  int h = nt * 16 + (lane & 15);
  P[idx] = (__bf16)W[((size_t)(k * 256 + h)) * 768 + d];
}
__global__ void k_pack_wx(const float* __restrict__ W, __bf16* __restrict__ P) {
  int idx = blockIdx.x * 256 + threadIdx.x;
  if (idx >= 7 * 16 * 10 * 512) return;
  int i = idx & 15, lane = (idx >> 4) & 31;
  int t = idx >> 9;
  int kt = t % 10, tc = t / 10, nt = tc & 15, k = tc >> 4;
  int d = kt * 32 + (i >> 3) * 16 + (lane >> 4) * 8 + (i & 7);
  int h = nt * 16 + (lane & 15);
  P[idx] = (d < 300) ? (__bf16)W[((size_t)(k * 256 + h)) * 300 + d] : (__bf16)0.f;
}
__global__ void k_sbf(const float* __restrict__ s, __bf16* __restrict__ sb) {
  int idx = blockIdx.x * 256 + threadIdx.x;
  if (idx >= 16384 * 320) return;
  int r = idx / 320, c = idx - r * 320;
  sb[idx] = (c < 300) ? (__bf16)s[(size_t)r * 300 + c] : (__bf16)0.f;
}
__global__ void k_hsbf(const float* __restrict__ hs, __bf16* __restrict__ hb) {
  int idx = blockIdx.x * 256 + threadIdx.x;
  if (idx < 32 * 514 * 256) hb[idx] = (__bf16)hs[idx];
}

// ---- per-step small kernels -------------------------------------------------
__global__ __launch_bounds__(256) void k_avg(const float* __restrict__ hs, const float* __restrict__ cs,
                                             float* __restrict__ havg, float* __restrict__ cgp) {
  int b = blockIdx.x, h = threadIdx.x;
  float s1 = 0.f, s2 = 0.f;
  for (int p = 0; p < 514; ++p) {
    size_t o = ((size_t)b * 514 + p) * 256 + h;
    s1 += hs[o]; s2 += cs[o];
  }
  havg[b * 256 + h] = s1 * (1.f / 514.f);
  cgp[b * 256 + h]  = s2 * (1.f / 514.f);
}

__global__ __launch_bounds__(256) void k_gates(
    const float* __restrict__ g, const float* __restrict__ havg,
    const float* __restrict__ Wfgg, const float* __restrict__ bfgg,
    const float* __restrict__ Wfgu, const float* __restrict__ bfgu,
    const float* __restrict__ Wog,  const float* __restrict__ bog,
    const float* __restrict__ Wou,  const float* __restrict__ bou,
    const float* __restrict__ Wfig, const float* __restrict__ bfig,
    float* __restrict__ fg, float* __restrict__ o, float* __restrict__ figb) {
  __shared__ float lg[256], lu[256], red[256];
  int b = blockIdx.x, h = threadIdx.x;
  lg[h] = g[b * 256 + h]; lu[h] = havg[b * 256 + h];
  __syncthreads();
  float d1 = bfgg[h], d2 = bfgu[h], d3 = bog[h], d4 = bou[h], d5 = bfig[h];
  for (int e = 0; e < 256; ++e) {
    float ge = lg[e], ue = lu[e];
    d1 += Wfgg[h * 256 + e] * ge;
    d2 += Wfgu[h * 256 + e] * ue;
    d3 += Wog [h * 256 + e] * ge;
    d4 += Wou [h * 256 + e] * ue;
    d5 += Wfig[h * 256 + e] * ge;
  }
  float sfg = sigf(d1 + d2);
  float ev = __expf(sfg);
  red[h] = ev; __syncthreads();
  for (int s = 128; s > 0; s >>= 1) { if (h < s) red[h] += red[h + s]; __syncthreads(); }
  fg[b * 256 + h]   = ev / red[0];
  o[b * 256 + h]    = sigf(d3 + d4);
  figb[b * 256 + h] = d5;
}

__global__ __launch_bounds__(256) void k_fi(
    const float* __restrict__ hs, const float* __restrict__ cs,
    const float* __restrict__ figb, const float* __restrict__ Wfiu,
    const float* __restrict__ bfiu, float* __restrict__ cg) {
  __shared__ float lhs[8 * 256];
  __shared__ float sval[8 * 256];
  __shared__ float rowsum[8];
  int b = blockIdx.x, p0 = blockIdx.y * 8;
  int t = threadIdx.x, lane = t & 31, wv = t >> 5;
  int nr = 514 - p0; if (nr > 8) nr = 8;
  #pragma unroll
  for (int j = 0; j < 8; ++j)
    lhs[j * 256 + t] = (j < nr) ? hs[((size_t)b * 514 + p0 + j) * 256 + t] : 0.f;
  __syncthreads();
  float base = figb[b * 256 + t] + bfiu[t];
  float av[8];
  #pragma unroll
  for (int j = 0; j < 8; ++j) av[j] = base;
  for (int e = 0; e < 256; ++e) {
    float w = Wfiu[(size_t)t * 256 + e];
    #pragma unroll
    for (int j = 0; j < 8; ++j) av[j] += w * lhs[j * 256 + e];
  }
  #pragma unroll
  for (int j = 0; j < 8; ++j) sval[j * 256 + t] = __expf(sigf(av[j]));
  __syncthreads();
  { // wave wv reduces row wv
    float s = 0.f;
    #pragma unroll
    for (int i = 0; i < 8; ++i) s += sval[wv * 256 + i * 32 + lane];
    #pragma unroll
    for (int off = 16; off; off >>= 1) s += __shfl_xor(s, off);
    if (lane == 0) rowsum[wv] = s;
  }
  __syncthreads();
  float contrib = 0.f;
  for (int j = 0; j < nr; ++j)
    contrib += (sval[j * 256 + t] / rowsum[j]) * cs[((size_t)b * 514 + p0 + j) * 256 + t];
  atomicAdd(&cg[b * 256 + t], contrib);
}

__global__ void k_gupd(const float* __restrict__ fg, const float* __restrict__ o,
                       const float* __restrict__ cgp, float* __restrict__ cg, float* __restrict__ g) {
  int i = blockIdx.x * 256 + threadIdx.x; // 8192 threads
  float c = cg[i] + fg[i] * cgp[i];
  cg[i] = c;
  g[i] = o[i] * tanhf(c);
}

__global__ __launch_bounds__(256) void k_gterm(const float* __restrict__ g,
                                               const float* __restrict__ Wg, float* __restrict__ gt) {
  __shared__ float lg[256];
  int kb = blockIdx.x, k = kb >> 5, b = kb & 31;
  int h = threadIdx.x;
  lg[h] = g[b * 256 + h]; __syncthreads();
  float d = 0.f;
  const float* w = Wg + ((size_t)(k * 256 + h)) * 256;
  for (int e = 0; e < 256; ++e) d += w[e] * lg[e];
  gt[((size_t)k * 32 + b) * 256 + h] = d;
}

// ---- fused WMMA GEMM (7 gates, K=768 hs + K=320 sents) + softmax + cell -----
__global__ __launch_bounds__(256) void k_fused(
    const __bf16* __restrict__ hsbf, const __bf16* __restrict__ sbf,
    const __bf16* __restrict__ pWep, const __bf16* __restrict__ pWx,
    const float* __restrict__ gterm, const float* __restrict__ cg,
    const float* __restrict__ b_ep, const float* __restrict__ b_x, const float* __restrict__ b_g,
    const float* __restrict__ cs_cur, float* __restrict__ hs_n, float* __restrict__ cs_n) {
  __shared__ float accL[7][16][256]; // 112 KB < 320 KB WGP LDS
  const int tid = threadIdx.x, lane = tid & 31, wv = tid >> 5;
  const int r0 = blockIdx.x * 16, b = r0 >> 9, l0 = r0 & 511;
  const int m = lane & 15;
  const __bf16* arow1 = hsbf + ((size_t)(b * 514 + l0 + m)) * 256; // ep rows overlap: stride 256, span 768
  const __bf16* arow2 = sbf + ((size_t)(r0 + m)) * 320;            // sents rows, K padded to 320
  const int tc0 = wv * 14; // 8 waves x 14 = 112 = 7 gates x 16 ntiles
  v8f acc[14];
  v8f z = {0.f, 0.f, 0.f, 0.f, 0.f, 0.f, 0.f, 0.f};
  #pragma unroll
  for (int j = 0; j < 14; ++j) acc[j] = z;
  for (int kt = 0; kt < 24; ++kt) {
    v16bf a = load_a(arow1 + kt * 32, lane);
    #pragma unroll
    for (int j = 0; j < 14; ++j) {
      const __bf16* bp = pWep + (((size_t)(tc0 + j)) * 24 + kt) * 512 + lane * 16;
      v16bf bb = load_b(bp);
      acc[j] = __builtin_amdgcn_wmma_f32_16x16x32_bf16(false, a, false, bb, (short)0, acc[j], false, false);
    }
  }
  for (int kt = 0; kt < 10; ++kt) {
    v16bf a = load_a(arow2 + kt * 32, lane);
    #pragma unroll
    for (int j = 0; j < 14; ++j) {
      const __bf16* bp = pWx + (((size_t)(tc0 + j)) * 10 + kt) * 512 + lane * 16;
      v16bf bb = load_b(bp);
      acc[j] = __builtin_amdgcn_wmma_f32_16x16x32_bf16(false, a, false, bb, (short)0, acc[j], false, false);
    }
  }
  { // C layout: VGPR e -> row (lane>>4)*8+e, col lane&15
    const int nl = lane & 15, mb = (lane >> 4) * 8;
    #pragma unroll
    for (int j = 0; j < 14; ++j) {
      int tc = tc0 + j, g_ = tc >> 4, nt = tc & 15;
      #pragma unroll
      for (int e = 0; e < 8; ++e)
        accL[g_][mb + e][nt * 16 + nl] = acc[j][e];
    }
  }
  __syncthreads();
  // epilogue: each wave handles 2 rows; lane covers h = i*32+lane
  const float* csb = cs_cur + ((size_t)b * 514) * 256;
  for (int rr = 0; rr < 2; ++rr) {
    const int mrow = wv * 2 + rr;
    const int l = l0 + mrow;
    float sv[5][8], ov[8], uv[8];
    float sm[5] = {0.f, 0.f, 0.f, 0.f, 0.f};
    #pragma unroll
    for (int i = 0; i < 8; ++i) {
      const int h = i * 32 + lane;
      #pragma unroll
      for (int k = 0; k < 7; ++k) {
        float v = accL[k][mrow][h] + gterm[((size_t)k * 32 + b) * 256 + h]
                + b_ep[k * 256 + h] + b_x[k * 256 + h] + b_g[k * 256 + h];
        if (k < 5)      { float s = sigf(v); sv[k][i] = s; sm[k] += __expf(s); }
        else if (k == 5) ov[i] = sigf(v);
        else             uv[i] = tanhf(v);
      }
    }
    #pragma unroll
    for (int k = 0; k < 5; ++k) { // wave32 cross-lane logsumexp
      float s = sm[k];
      #pragma unroll
      for (int off = 16; off; off >>= 1) s += __shfl_xor(s, off);
      sm[k] = __logf(s);
    }
    #pragma unroll
    for (int i = 0; i < 8; ++i) {
      const int h = i * 32 + lane;
      float it = sv[0][i] - sm[0], lt = sv[1][i] - sm[1], rt = sv[2][i] - sm[2],
            ft = sv[3][i] - sm[3], st = sv[4][i] - sm[4];
      float ct = lt * csb[(size_t)l * 256 + h] + ft * csb[(size_t)(l + 1) * 256 + h]
               + rt * csb[(size_t)(l + 2) * 256 + h] + st * cg[b * 256 + h] + it * uv[i];
      float ht = ov[i] * tanhf(ct);
      size_t o = ((size_t)b * 514 + (l + 1)) * 256 + h;
      hs_n[o] = ht; cs_n[o] = ct;
    }
  }
  if (l0 == 0)   { size_t o = ((size_t)b * 514) * 256 + tid;       hs_n[o] = 0.f; cs_n[o] = 0.f; }
  if (l0 == 496) { size_t o = ((size_t)b * 514 + 513) * 256 + tid; hs_n[o] = 0.f; cs_n[o] = 0.f; }
}

__global__ void k_final(const float* __restrict__ hs, const float* __restrict__ g, float* __restrict__ out) {
  int idx = blockIdx.x * 256 + threadIdx.x;
  const int NBLH = 32 * 512 * 256;
  if (idx < NBLH) {
    int b = idx >> 17, rem = idx & 131071;
    int l = rem >> 8, h = rem & 255;
    out[idx] = hs[((size_t)b * 514 + l + 1) * 256 + h];
  } else if (idx < NBLH + 32 * 256) {
    out[idx] = g[idx - NBLH];
  }
}

extern "C" void kernel_launch(void* const* d_in, const int* in_sizes, int n_in,
                              void* d_out, int out_size, void* d_ws, size_t ws_size,
                              hipStream_t stream) {
  const float* sents = (const float*)d_in[0];
  const float* hs0 = (const float*)d_in[2];
  const float* cs0 = (const float*)d_in[3];
  const float* g0  = (const float*)d_in[4];
  const float* W_ep = (const float*)d_in[5];
  const float* b_ep = (const float*)d_in[6];
  const float* W_x  = (const float*)d_in[7];
  const float* b_x  = (const float*)d_in[8];
  const float* W_g  = (const float*)d_in[9];
  const float* b_g  = (const float*)d_in[10];
  const float* W_fgg = (const float*)d_in[11]; const float* b_fgg = (const float*)d_in[12];
  const float* W_fgu = (const float*)d_in[13]; const float* b_fgu = (const float*)d_in[14];
  const float* W_fig = (const float*)d_in[15]; const float* b_fig = (const float*)d_in[16];
  const float* W_fiu = (const float*)d_in[17]; const float* b_fiu = (const float*)d_in[18];
  const float* W_og  = (const float*)d_in[19]; const float* b_og  = (const float*)d_in[20];
  const float* W_ou  = (const float*)d_in[21]; const float* b_ou  = (const float*)d_in[22];

  char* base = (char*)d_ws;
  size_t off = 0;
  auto alloc = [&](size_t bytes) -> char* {
    off = (off + 255) & ~(size_t)255;
    char* p = base + off; off += bytes; return p;
  };
  const size_t HS_B = (size_t)32 * 514 * 256 * 4;
  float* hsA = (float*)alloc(HS_B); float* hsB = (float*)alloc(HS_B);
  float* csA = (float*)alloc(HS_B); float* csB = (float*)alloc(HS_B);
  __bf16* hsbf = (__bf16*)alloc((size_t)32 * 514 * 256 * 2);
  __bf16* sbf  = (__bf16*)alloc((size_t)16384 * 320 * 2);
  __bf16* pWep = (__bf16*)alloc((size_t)7 * 16 * 24 * 512 * 2);
  __bf16* pWx  = (__bf16*)alloc((size_t)7 * 16 * 10 * 512 * 2);
  float* gbuf = (float*)alloc(32 * 256 * 4);
  float* havg = (float*)alloc(32 * 256 * 4);
  float* cgp  = (float*)alloc(32 * 256 * 4);
  float* cgv  = (float*)alloc(32 * 256 * 4);
  float* fgv  = (float*)alloc(32 * 256 * 4);
  float* ov   = (float*)alloc(32 * 256 * 4);
  float* figb = (float*)alloc(32 * 256 * 4);
  float* gterm = (float*)alloc((size_t)7 * 32 * 256 * 4);
  if (off > ws_size) return;

  hipMemcpyAsync(hsA, hs0, HS_B, hipMemcpyDeviceToDevice, stream);
  hipMemcpyAsync(csA, cs0, HS_B, hipMemcpyDeviceToDevice, stream);
  hipMemcpyAsync(gbuf, g0, 32 * 256 * 4, hipMemcpyDeviceToDevice, stream);

  k_pack_wep<<<(7 * 16 * 24 * 512 + 255) / 256, 256, 0, stream>>>(W_ep, pWep);
  k_pack_wx <<<(7 * 16 * 10 * 512 + 255) / 256, 256, 0, stream>>>(W_x, pWx);
  k_sbf     <<<(16384 * 320 + 255) / 256, 256, 0, stream>>>(sents, sbf);

  float* hs_c = hsA; float* cs_c = csA; float* hs_n = hsB; float* cs_n = csB;
  for (int t = 0; t < 6; ++t) {
    k_avg<<<32, 256, 0, stream>>>(hs_c, cs_c, havg, cgp);
    k_gates<<<32, 256, 0, stream>>>(gbuf, havg, W_fgg, b_fgg, W_fgu, b_fgu,
                                    W_og, b_og, W_ou, b_ou, W_fig, b_fig, fgv, ov, figb);
    hipMemsetAsync(cgv, 0, 32 * 256 * 4, stream);
    k_fi<<<dim3(32, 65), 256, 0, stream>>>(hs_c, cs_c, figb, W_fiu, b_fiu, cgv);
    k_gupd<<<32, 256, 0, stream>>>(fgv, ov, cgp, cgv, gbuf);
    k_gterm<<<224, 256, 0, stream>>>(gbuf, W_g, gterm);
    k_hsbf<<<(32 * 514 * 256 + 255) / 256, 256, 0, stream>>>(hs_c, hsbf);
    k_fused<<<1024, 256, 0, stream>>>(hsbf, sbf, pWep, pWx, gterm, cgv,
                                      b_ep, b_x, b_g, cs_c, hs_n, cs_n);
    float* th = hs_c; hs_c = hs_n; hs_n = th;
    float* tc = cs_c; cs_c = cs_n; cs_n = tc;
  }
  k_final<<<(32 * 512 * 256 + 32 * 256 + 255) / 256, 256, 0, stream>>>(hs_c, gbuf, (float*)d_out);
}
